// RecTransformer_74002286510735
// MI455X (gfx1250) — compile-verified, matrix operand order checked by reading, not run on hardware
//
#include <hip/hip_runtime.h>
#include <hip/hip_bf16.h>

// ---------------- model constants ----------------
#define BB    32
#define TT    200
#define DD    512
#define HH    8      // query heads
#define KVH   2      // kv heads
#define REP   4      // HH/KVH
#define HD    64     // head dim
#define LL    4      // layers
#define VV    20001
#define KVD   128    // KVH*HD
#define MROWS (BB*TT)        // 6400
#define NPADV 20096          // V padded to multiple of 128

typedef __attribute__((ext_vector_type(16))) _Float16 v16h_t;
typedef __attribute__((ext_vector_type(8)))  _Float16 v8h_t;
typedef __attribute__((ext_vector_type(8)))  float    v8f_t;

// ---------------------------------------------------------------
// Embedding: h[b,t,:] = item_emb[x[b,t],:] + pos_emb[t,:]
// writes f32 (residual path) and f16 (WMMA input) copies
// ---------------------------------------------------------------
__global__ void embed_kernel(const int* __restrict__ x,
                             const float* __restrict__ item_emb,
                             const float* __restrict__ pos_emb,
                             float* __restrict__ h32,
                             _Float16* __restrict__ h16) {
    int idx = blockIdx.x * blockDim.x + threadIdx.x;   // < MROWS*DD
    int d   = idx & (DD - 1);
    int bt  = idx >> 9;                                // DD == 512
    int t   = bt % TT;
    int tok = x[bt];
    float v = item_emb[(size_t)tok * DD + d] + pos_emb[(size_t)t * DD + d];
    h32[idx] = v;
    h16[idx] = (_Float16)v;
}

// ---------------------------------------------------------------
// Weight transpose + f32->f16 convert (+ zero pad rows):
//   W  [K, N] f32  (row-major, as in reference h @ W)
//   Wt [Npad, K] f16, Wt[n][k] = W[k][n], zero for n >= N
// ---------------------------------------------------------------
__global__ void transpose_w_kernel(const float* __restrict__ W,
                                   _Float16* __restrict__ Wt,
                                   int K, int N, int Npad) {
    size_t idx = (size_t)blockIdx.x * blockDim.x + threadIdx.x;
    size_t total = (size_t)Npad * K;
    if (idx >= total) return;
    int n = (int)(idx / K);
    int k = (int)(idx % K);
    Wt[idx] = (n < N) ? (_Float16)W[(size_t)k * N + n] : (_Float16)0.0f;
}

// ---------------------------------------------------------------
// A/B fragment load per ISA 7.12.2 (16-bit A-matrix 16x32):
//   lane l: row = l&15, kb = (l>>4)*8; elems 0..7 -> K=kb..kb+7,
//   elems 8..15 -> K=kb+16..kb+23 (two contiguous 16B loads).
// ---------------------------------------------------------------
__device__ __forceinline__ v16h_t load_frag(const _Float16* p) {
    v8h_t lo = *(const v8h_t*)p;
    v8h_t hi = *(const v8h_t*)(p + 16);
    return __builtin_shufflevector(lo, hi, 0,1,2,3,4,5,6,7,8,9,10,11,12,13,14,15);
}

// C/D layout (ISA 7.12.2): VGPR r -> M = r + 8*(lane>>4), N = lane&15
__device__ __forceinline__ void store_tile(float* __restrict__ C, v8f_t acc,
                                           const float* __restrict__ bias,
                                           int m0, int n, int Nreal, int ldc) {
    if (n >= Nreal) return;
    float bia = bias ? bias[n] : 0.0f;
#pragma unroll
    for (int r = 0; r < 8; ++r)
        C[(size_t)(m0 + r) * ldc + n] = acc[r] + bia;
}

// ---------------------------------------------------------------
// WMMA GEMM:  C[M, ldc] (+bias) = A[M,K]f16 @ Bt[Npad,K]f16^T
// Register-blocked 2x2: each wave owns a 32x32 output block
// (4 v_wmma_f32_16x16x32_f16 per K-step, each A/B fragment reused
// twice -> 32 B/lane (1 KB/wave) of L2 traffic per WMMA). 4 waves/block -> 32x128
// per workgroup. grid = (M/32, Npad/128), block = 128.
// ---------------------------------------------------------------
__global__ void gemm_wmma_kernel(const _Float16* __restrict__ A,
                                 const _Float16* __restrict__ Bt,
                                 const float* __restrict__ bias,
                                 float* __restrict__ C,
                                 int K, int Nreal, int ldc) {
    const int lane = threadIdx.x & 31;
    const int wave = threadIdx.x >> 5;
    const int tm = blockIdx.x * 32;               // 2 m-tiles
    const int tn = blockIdx.y * 128 + wave * 32;  // 2 n-tiles
    const int r16 = lane & 15;
    const int kb  = (lane >> 4) * 8;

    const _Float16* ap0 = A  + (size_t)(tm + r16) * K + kb;
    const _Float16* ap1 = ap0 + (size_t)16 * K;
    const _Float16* bp0 = Bt + (size_t)(tn + r16) * K + kb;
    const _Float16* bp1 = bp0 + (size_t)16 * K;

    v8f_t acc00 = {}, acc01 = {}, acc10 = {}, acc11 = {};
    for (int k0 = 0; k0 < K; k0 += 32) {
        v16h_t a0 = load_frag(ap0 + k0);
        v16h_t a1 = load_frag(ap1 + k0);
        v16h_t b0 = load_frag(bp0 + k0);
        v16h_t b1 = load_frag(bp1 + k0);
        acc00 = __builtin_amdgcn_wmma_f32_16x16x32_f16(false, a0, false, b0, (short)0, acc00, false, false);
        acc01 = __builtin_amdgcn_wmma_f32_16x16x32_f16(false, a0, false, b1, (short)0, acc01, false, false);
        acc10 = __builtin_amdgcn_wmma_f32_16x16x32_f16(false, a1, false, b0, (short)0, acc10, false, false);
        acc11 = __builtin_amdgcn_wmma_f32_16x16x32_f16(false, a1, false, b1, (short)0, acc11, false, false);
    }

    const int m0 = tm + ((lane >> 4) << 3);
    const int n0 = tn + r16;
    store_tile(C, acc00, bias, m0,      n0,      Nreal, ldc);
    store_tile(C, acc01, bias, m0,      n0 + 16, Nreal, ldc);
    store_tile(C, acc10, bias, m0 + 16, n0,      Nreal, ldc);
    store_tile(C, acc11, bias, m0 + 16, n0 + 16, Nreal, ldc);
}

// ---------------------------------------------------------------
// Causal GQA attention, one block per (b, head), thread = query row.
// Two-pass online softmax: pass 1 max+sum, pass 2 weighted V accum.
// q32 [MROWS, DD] (head h slice at h*64), k32/v32 [MROWS, KVD]
// (group g slice at g*64). Writes o16 [MROWS, DD] f16.
// ---------------------------------------------------------------
__global__ void attn_kernel(const float* __restrict__ q32,
                            const float* __restrict__ k32,
                            const float* __restrict__ v32,
                            _Float16* __restrict__ o16) {
    const int head = blockIdx.x;     // 0..BB*HH-1
    const int b = head >> 3;         // / HH
    const int h = head & 7;
    const int g = h >> 2;            // h = g*REP + r
    const int t = threadIdx.x;
    if (t >= TT) return;

    const float scale = 0.125f;      // 1/sqrt(64)
    const float* qp = q32 + (size_t)(b * TT + t) * DD + h * HD;
    float qr[HD];
#pragma unroll
    for (int i = 0; i < HD; ++i) qr[i] = qp[i] * scale;

    const float* kbase = k32 + (size_t)(b * TT) * KVD + g * HD;
    const float* vbase = v32 + (size_t)(b * TT) * KVD + g * HD;

    // pass 1: running max + sum of exp
    float mx = -1e30f, sum = 0.0f;
    for (int s = 0; s <= t; ++s) {
        const float* kp = kbase + (size_t)s * KVD;
        float d = 0.0f;
#pragma unroll
        for (int i = 0; i < HD; ++i) d += qr[i] * kp[i];
        if (d > mx) {
            sum = sum * __expf(mx - d) + 1.0f;
            mx = d;
        } else {
            sum += __expf(d - mx);
        }
    }
    float inv = 1.0f / sum;

    // pass 2: output accumulation
    float accv[HD];
#pragma unroll
    for (int i = 0; i < HD; ++i) accv[i] = 0.0f;
    for (int s = 0; s <= t; ++s) {
        const float* kp = kbase + (size_t)s * KVD;
        float d = 0.0f;
#pragma unroll
        for (int i = 0; i < HD; ++i) d += qr[i] * kp[i];
        float w = __expf(d - mx) * inv;
        const float* vp = vbase + (size_t)s * KVD;
#pragma unroll
        for (int i = 0; i < HD; ++i) accv[i] += w * vp[i];
    }

    _Float16* op = o16 + (size_t)(b * TT + t) * DD + h * HD;
#pragma unroll
    for (int i = 0; i < HD; ++i) op[i] = (_Float16)accv[i];
}

// ---------------------------------------------------------------
// h = LayerNorm(h + p) * gamma + beta  (in place on h32, h16)
// one 256-thread block per row of 512
// ---------------------------------------------------------------
__global__ void add_ln_kernel(float* __restrict__ h32,
                              _Float16* __restrict__ h16,
                              const float* __restrict__ p,
                              const float* __restrict__ gamma,
                              const float* __restrict__ beta) {
    const int row = blockIdx.x;
    const int tid = threadIdx.x;      // 256
    __shared__ float red[256];
    size_t base = (size_t)row * DD;

    float x0 = h32[base + tid]       + p[base + tid];
    float x1 = h32[base + 256 + tid] + p[base + 256 + tid];

    red[tid] = x0 + x1;
    __syncthreads();
    for (int off = 128; off > 0; off >>= 1) {
        if (tid < off) red[tid] += red[tid + off];
        __syncthreads();
    }
    float mu = red[0] * (1.0f / DD);
    __syncthreads();

    float d0 = x0 - mu, d1 = x1 - mu;
    red[tid] = d0 * d0 + d1 * d1;
    __syncthreads();
    for (int off = 128; off > 0; off >>= 1) {
        if (tid < off) red[tid] += red[tid + off];
        __syncthreads();
    }
    float rstd = rsqrtf(red[0] * (1.0f / DD) + 1e-5f);

    float y0 = d0 * rstd * gamma[tid]       + beta[tid];
    float y1 = d1 * rstd * gamma[tid + 256] + beta[tid + 256];
    h32[base + tid]       = y0;
    h32[base + 256 + tid] = y1;
    h16[base + tid]       = (_Float16)y0;
    h16[base + 256 + tid] = (_Float16)y1;
}

// ---------------------------------------------------------------
extern "C" void kernel_launch(void* const* d_in, const int* in_sizes, int n_in,
                              void* d_out, int out_size, void* d_ws, size_t ws_size,
                              hipStream_t stream) {
    const int*   x        = (const int*)  d_in[0];
    const float* item_emb = (const float*)d_in[1];
    const float* pos_emb  = (const float*)d_in[2];
    const float* qw       = (const float*)d_in[3];
    const float* qb       = (const float*)d_in[4];
    const float* kw       = (const float*)d_in[5];
    const float* kb_      = (const float*)d_in[6];
    const float* vw       = (const float*)d_in[7];
    const float* vb_      = (const float*)d_in[8];
    const float* ow       = (const float*)d_in[9];
    const float* ob       = (const float*)d_in[10];
    const float* gamma    = (const float*)d_in[11];
    const float* beta     = (const float*)d_in[12];
    const float* fcw      = (const float*)d_in[13];
    const float* fcb      = (const float*)d_in[14];
    float* out = (float*)d_out;

    // -------- workspace carve (256B aligned) --------
    char* wsp = (char*)d_ws;
    auto carve = [&](size_t bytes) -> char* {
        char* p = wsp;
        wsp += (bytes + 255) & ~(size_t)255;
        return p;
    };
    float*    h32  = (float*)   carve((size_t)MROWS * DD * 4);
    _Float16* h16  = (_Float16*)carve((size_t)MROWS * DD * 2);
    float*    q32  = (float*)   carve((size_t)MROWS * DD * 4);
    float*    k32  = (float*)   carve((size_t)MROWS * KVD * 4);
    float*    v32  = (float*)   carve((size_t)MROWS * KVD * 4);
    _Float16* o16  = (_Float16*)carve((size_t)MROWS * DD * 2);
    float*    p32  = (float*)   carve((size_t)MROWS * DD * 4);
    _Float16* qwt  = (_Float16*)carve((size_t)LL * DD * DD * 2);
    _Float16* kwt  = (_Float16*)carve((size_t)LL * KVD * DD * 2);
    _Float16* vwt  = (_Float16*)carve((size_t)LL * KVD * DD * 2);
    _Float16* owt  = (_Float16*)carve((size_t)LL * DD * DD * 2);
    _Float16* fcwt = (_Float16*)carve((size_t)NPADV * DD * 2);

    // -------- weight transpose / f16 convert --------
    for (int l = 0; l < LL; ++l) {
        transpose_w_kernel<<<(DD * DD + 255) / 256, 256, 0, stream>>>(
            qw + (size_t)l * DD * DD, qwt + (size_t)l * DD * DD, DD, DD, DD);
        transpose_w_kernel<<<(KVD * DD + 255) / 256, 256, 0, stream>>>(
            kw + (size_t)l * DD * KVD, kwt + (size_t)l * KVD * DD, DD, KVD, KVD);
        transpose_w_kernel<<<(KVD * DD + 255) / 256, 256, 0, stream>>>(
            vw + (size_t)l * DD * KVD, vwt + (size_t)l * KVD * DD, DD, KVD, KVD);
        transpose_w_kernel<<<(DD * DD + 255) / 256, 256, 0, stream>>>(
            ow + (size_t)l * DD * DD, owt + (size_t)l * DD * DD, DD, DD, DD);
    }
    transpose_w_kernel<<<((size_t)NPADV * DD + 255) / 256, 256, 0, stream>>>(
        fcw, fcwt, DD, VV, NPADV);

    // -------- embedding --------
    embed_kernel<<<(MROWS * DD) / 256, 256, 0, stream>>>(x, item_emb, pos_emb, h32, h16);

    // -------- transformer layers --------
    const int MB = MROWS / 32;   // 200 row-blocks
    for (int l = 0; l < LL; ++l) {
        gemm_wmma_kernel<<<dim3(MB, DD / 128), 128, 0, stream>>>(
            h16, qwt + (size_t)l * DD * DD, qb + (size_t)l * DD, q32, DD, DD, DD);
        gemm_wmma_kernel<<<dim3(MB, KVD / 128), 128, 0, stream>>>(
            h16, kwt + (size_t)l * KVD * DD, kb_ + (size_t)l * KVD, k32, DD, KVD, KVD);
        gemm_wmma_kernel<<<dim3(MB, KVD / 128), 128, 0, stream>>>(
            h16, vwt + (size_t)l * KVD * DD, vb_ + (size_t)l * KVD, v32, DD, KVD, KVD);
        attn_kernel<<<BB * HH, 256, 0, stream>>>(q32, k32, v32, o16);
        gemm_wmma_kernel<<<dim3(MB, DD / 128), 128, 0, stream>>>(
            o16, owt + (size_t)l * DD * DD, ob + (size_t)l * DD, p32, DD, DD, DD);
        add_ln_kernel<<<MROWS, 256, 0, stream>>>(h32, h16, p32, gamma, beta);
    }

    // -------- final projection to vocabulary (dominant GEMM) --------
    gemm_wmma_kernel<<<dim3(MB, NPADV / 128), 128, 0, stream>>>(
        h16, fcwt, fcb, out, DD, VV, VV);
}
